// GAE_1108101562624
// MI455X (gfx1250) — matrix operand order repaired and checked by reading, not compiled, not run on hardware
//
#include <hip/hip_runtime.h>

// ---------------------------------------------------------------------------
// MI455X (gfx1250) implementation.
// Dominant work: pair-MLP (100000x1000x512 bf16 WMMA GEMM, fused with the
// factored broadcast-add A1[a]+O1[o]+b1, LN, relu, LN) and the final
// 256x100000x512 WMMA GEMM. All GEMMs run through v_wmma_f32_16x16x32_bf16.
// GEMM A-tiles are staged to LDS with gfx1250 async copies
// (global_load_async_to_lds_b128 + s_wait_asynccnt).
// ---------------------------------------------------------------------------

typedef __attribute__((ext_vector_type(16))) __bf16   v16bf;
typedef __attribute__((ext_vector_type(8)))  float    v8f;
typedef __attribute__((ext_vector_type(4)))  unsigned int u32x4;

union FragU { u32x4 q[2]; v16bf v; };

__device__ inline v8f wmma_bf16(v16bf a, v16bf b, v8f c) {
    return __builtin_amdgcn_wmma_f32_16x16x32_bf16(
        /*neg_a=*/false, a, /*neg_b=*/false, b,
        /*c_mod=*/(short)0, c, /*reuse_a=*/false, /*reuse_b=*/false);
}

__device__ inline unsigned short f2bf(float f) {
    unsigned int u = __float_as_uint(f);
    u = u + 0x7FFFu + ((u >> 16) & 1u);   // round-to-nearest-even
    return (unsigned short)(u >> 16);
}

__device__ inline float wred(float v) {
    #pragma unroll
    for (int o = 16; o; o >>= 1) v += __shfl_xor(v, o, 32);
    return v;
}

// async 16B copy: global -> LDS, tracked by ASYNCcnt
__device__ inline void async_cp_b128(unsigned lds_byte_off, const void* gaddr) {
    asm volatile("global_load_async_to_lds_b128 %0, %1, off"
                 :: "v"(lds_byte_off), "v"(gaddr) : "memory");
}
__device__ inline void wait_async0() {
    asm volatile("s_wait_asynccnt 0x0" ::: "memory");
}

// ---------------------------------------------------------------------------
// small utility kernels
// ---------------------------------------------------------------------------
__global__ void zero_f32(float* p, int n) {
    int i = blockIdx.x * 256 + threadIdx.x;
    if (i < n) p[i] = 0.f;
}

__global__ void seg_count(const int* dst, float* cnt, int E) {
    int e = blockIdx.x * 256 + threadIdx.x;
    if (e < E) atomicAdd(&cnt[dst[e]], 1.0f);
}

__global__ void seg_accum(const float* x, const int* src, const int* dst,
                          float* agg, int E, int F) {
    int idx = blockIdx.x * 256 + threadIdx.x;
    if (idx >= E * F) return;
    int e = idx / F, f = idx % F;
    atomicAdd(&agg[(long)dst[e] * F + f], x[(long)src[e] * F + f]);
}

__global__ void seg_mean(const float* agg, const float* cnt, float* mean,
                         int n, int F) {
    int idx = blockIdx.x * 256 + threadIdx.x;
    if (idx >= n) return;
    mean[idx] = agg[idx] / fmaxf(cnt[idx / F], 1.0f);
}

// W[K][N] (row stride ldw) -> Wt[N][Kpad] bf16, zero-padded K (B operand)
__global__ void tconv(const float* W, unsigned short* Wt, int K, int N,
                      int ldw, int Kpad) {
    int idx = blockIdx.x * 256 + threadIdx.x;
    if (idx >= N * Kpad) return;
    int n = idx / Kpad, k = idx % Kpad;
    Wt[idx] = (k < K) ? f2bf(W[(long)k * ldw + n]) : (unsigned short)0;
}

// A[M][K] f32 -> Abf[Mpad][Kpad] bf16, zero padded rows/cols (A operand)
__global__ void convA(const float* A, unsigned short* Abf, int M, int Mpad,
                      int K, int Kpad) {
    int idx = blockIdx.x * 256 + threadIdx.x;
    if (idx >= Mpad * Kpad) return;
    int m = idx / Kpad, k = idx % Kpad;
    Abf[idx] = (m < M && k < K) ? f2bf(A[(long)m * K + k]) : (unsigned short)0;
}

// per-row LayerNorm (+optional relu), one wave per row
__global__ void ln_rows(const float* x, const float* g, const float* b,
                        float* y, int rows, int cols, int doRelu) {
    int w = threadIdx.x >> 5, lane = threadIdx.x & 31;
    int r = blockIdx.x * 8 + w;
    if (r >= rows) return;
    const float* xr = x + (long)r * cols;
    float s = 0.f, s2 = 0.f;
    for (int k = lane; k < cols; k += 32) { float v = xr[k]; s += v; s2 += v * v; }
    s = wred(s); s2 = wred(s2);
    float mu = s / cols, var = s2 / cols - mu * mu, rs = rsqrtf(var + 1e-5f);
    float* yr = y + (long)r * cols;
    for (int k = lane; k < cols; k += 32) {
        float v = (xr[k] - mu) * rs * g[k] + b[k];
        if (doRelu) v = fmaxf(v, 0.f);
        yr[k] = v;
    }
}

// ---------------------------------------------------------------------------
// Generic WMMA GEMM: C[M][ldc] (+)= Abf[Mpad][Kpad](bf16) * Bt[N][Kpad](bf16)^T
// block tile 64x128, 8 waves (4m x 2n), each wave 16x64 -> 4 wmma frags.
// A tile staged to LDS via async global->LDS b128 copies (ASYNCcnt).
// Abf must be padded so rows [m0, m0+64) are always readable.
// ---------------------------------------------------------------------------
template <bool ADD, bool RELU, bool BIAS>
__global__ void gemm_bf16t(const unsigned short* __restrict__ Abf,
                           const unsigned short* __restrict__ Bt,
                           const float* __restrict__ bias,
                           float* __restrict__ C,
                           int M, int N, int Kpad, int ldc) {
    __shared__ __align__(16) unsigned short sA[64 * 40];   // 64 rows x 32 bf16, pad->40
    const int tid = threadIdx.x, lane = tid & 31, w = tid >> 5;
    const int wm = w & 3, wn = w >> 2;
    const int m0 = blockIdx.x * 64, n0 = blockIdx.y * 128;
    const int g = lane >> 4, ln16 = lane & 15;

    // per-thread async copy assignment: 64 rows x 4 chunks of 16B
    const int arow = tid >> 2, ach = tid & 3;
    const unsigned ldsOff =
        (unsigned)(unsigned long long)(&sA[arow * 40 + ach * 8]);
    const unsigned short* agp = Abf + (long)(m0 + arow) * Kpad + ach * 8;

    v8f acc[4] = {};
    for (int kt = 0; kt < Kpad; kt += 32) {
        __syncthreads();                       // prior frag reads done
        async_cp_b128(ldsOff, agp + kt);       // 16B/thread -> 4KB tile
        wait_async0();
        __syncthreads();                       // tile visible to all waves
        FragU a;
        a.q[0] = *(const u32x4*)&sA[(wm * 16 + ln16) * 40 + g * 8];
        a.q[1] = *(const u32x4*)&sA[(wm * 16 + ln16) * 40 + 16 + g * 8];
        #pragma unroll
        for (int j = 0; j < 4; ++j) {
            int n = n0 + wn * 64 + j * 16 + ln16;
            FragU b;
            if (n < N) {
                const unsigned short* bp = Bt + (long)n * Kpad + kt + g * 8;
                b.q[0] = *(const u32x4*)bp;
                b.q[1] = *(const u32x4*)(bp + 16);
            } else {
                u32x4 zz = {0u, 0u, 0u, 0u};
                b.q[0] = zz; b.q[1] = zz;
            }
            acc[j] = wmma_bf16(a.v, b.v, acc[j]);
        }
    }
    #pragma unroll
    for (int j = 0; j < 4; ++j) {
        int n = n0 + wn * 64 + j * 16 + ln16;
        if (n >= N) continue;
        #pragma unroll
        for (int v = 0; v < 8; ++v) {
            int m = m0 + wm * 16 + g * 8 + v;
            if (m >= M) continue;
            float val = acc[j][v];
            if (BIAS) val += bias[n];
            long off = (long)m * ldc + n;
            if (ADD) val += C[off];
            if (RELU) val = fmaxf(val, 0.f);
            C[off] = val;
        }
    }
}

// ---------------------------------------------------------------------------
// Fused pair kernel: 16 pairs per block.
//   z = A1[a] + O1[o] (b1 folded into A1 GEMM bias) -> LN -> relu -> u (LDS)
//   raw = u @ p_W2 + b2 (WMMA, u resident in LDS bf16) -> LN -> bf16 pairN
// LDS: u[16][1032] bf16 (33024B) aliased with out[16][520] f32 (33280B)
// ---------------------------------------------------------------------------
__global__ void pair_fused(const float* __restrict__ A1,
                           const float* __restrict__ O1,
                           const float* __restrict__ g1,
                           const float* __restrict__ be1,
                           const unsigned short* __restrict__ W2t,  // [512][1024]
                           const float* __restrict__ b2,
                           const float* __restrict__ gn,
                           const float* __restrict__ bn,
                           unsigned short* __restrict__ pairN) {
    __shared__ __align__(16) unsigned char smem[16 * 520 * 4];
    unsigned short* U = (unsigned short*)smem;   // 16 x 1032 bf16
    float* OT = (float*)smem;                    // 16 x 520  f32

    const int tid = threadIdx.x, lane = tid & 31, w = tid >> 5;
    const int p0 = blockIdx.x * 16;
    const int g = lane >> 4, ln16 = lane & 15;

    // ---- phase A: broadcast add + LN + relu -> u (bf16, LDS) ----
    #pragma unroll
    for (int i = 0; i < 2; ++i) {
        int r = w * 2 + i;
        int p = p0 + r;
        int a = p / 500, o = p % 500;
        const float* ar = A1 + (long)a * 1000;   // already includes +b1
        const float* orow = O1 + (long)o * 1000;
        float s = 0.f, s2 = 0.f;
        for (int k = lane; k < 1000; k += 32) {
            float z = ar[k] + orow[k];
            s += z; s2 += z * z;
        }
        s = wred(s); s2 = wred(s2);
        float mu = s * 0.001f, var = s2 * 0.001f - mu * mu, rs = rsqrtf(var + 1e-5f);
        for (int k = lane; k < 1000; k += 32) {
            float z = ar[k] + orow[k];
            float y = fmaxf((z - mu) * rs * g1[k] + be1[k], 0.f);
            U[r * 1032 + k] = f2bf(y);
        }
        for (int k = 1000 + lane; k < 1032; k += 32) U[r * 1032 + k] = 0;
    }
    __syncthreads();

    // ---- phase B: u(16x1024) @ W2t^T -> 16x512; wave w owns cols [64w,64w+64) ----
    v8f acc[4] = {};
    for (int kt = 0; kt < 1024; kt += 32) {
        FragU a;
        a.q[0] = *(const u32x4*)&U[ln16 * 1032 + kt + g * 8];
        a.q[1] = *(const u32x4*)&U[ln16 * 1032 + kt + 16 + g * 8];
        #pragma unroll
        for (int j = 0; j < 4; ++j) {
            int n = w * 64 + j * 16 + ln16;
            const unsigned short* bp = W2t + (long)n * 1024 + kt + g * 8;
            FragU b;
            b.q[0] = *(const u32x4*)bp;
            b.q[1] = *(const u32x4*)(bp + 16);
            acc[j] = wmma_bf16(a.v, b.v, acc[j]);
        }
    }
    __syncthreads();   // everyone done reading U; safe to alias with OT
    #pragma unroll
    for (int j = 0; j < 4; ++j) {
        int col = w * 64 + j * 16 + ln16;
        #pragma unroll
        for (int v = 0; v < 8; ++v) {
            int m = g * 8 + v;
            OT[m * 520 + col] = acc[j][v] + b2[col];
        }
    }
    __syncthreads();

    // ---- phase C: LN over 512 cols, write bf16 pair_nodes ----
    #pragma unroll
    for (int i = 0; i < 2; ++i) {
        int r = w * 2 + i;
        int p = p0 + r;
        float s = 0.f, s2 = 0.f;
        for (int k = lane; k < 512; k += 32) { float v = OT[r * 520 + k]; s += v; s2 += v * v; }
        s = wred(s); s2 = wred(s2);
        float mu = s * (1.f / 512.f), var = s2 * (1.f / 512.f) - mu * mu;
        float rs = rsqrtf(var + 1e-5f);
        unsigned short* out = pairN + (long)p * 512;
        for (int k = lane; k < 512; k += 32)
            out[k] = f2bf((OT[r * 520 + k] - mu) * rs * gn[k] + bn[k]);
    }
}

// ---------------------------------------------------------------------------
// host launcher
// ---------------------------------------------------------------------------
static inline int cdiv(long a, long b) { return (int)((a + b - 1) / b); }

extern "C" void kernel_launch(void* const* d_in, const int* in_sizes, int n_in,
                              void* d_out, int out_size, void* d_ws, size_t ws_size,
                              hipStream_t stream) {
    (void)in_sizes; (void)n_in; (void)out_size; (void)ws_size;
    const int NATTRS = 200, NOBJS = 500, N = 700, E = 50000;
    const int HID = 2048, NODE_D = 512, B = 256, EMB = 512;
    const long OUT_NODES_OFF = (long)B * 100000;   // pair_pred then nodes

    const float* img        = (const float*)d_in[0];
    const float* node_feats = (const float*)d_in[1];
    const int*   src        = (const int*)d_in[2];
    const int*   dst        = src + E;
    const float* s1_Wl = (const float*)d_in[3];
    const float* s1_bl = (const float*)d_in[4];
    const float* s1_Wr = (const float*)d_in[5];
    const float* s2_Wl = (const float*)d_in[6];
    const float* s2_bl = (const float*)d_in[7];
    const float* s2_Wr = (const float*)d_in[8];
    const float* i_W1 = (const float*)d_in[9];   const float* i_b1 = (const float*)d_in[10];
    const float* i_g1 = (const float*)d_in[11];  const float* i_be1 = (const float*)d_in[12];
    const float* i_W2 = (const float*)d_in[13];  const float* i_b2 = (const float*)d_in[14];
    const float* i_g2 = (const float*)d_in[15];  const float* i_be2 = (const float*)d_in[16];
    const float* i_W3 = (const float*)d_in[17];  const float* i_b3 = (const float*)d_in[18];
    const float* i_gn = (const float*)d_in[19];  const float* i_bn = (const float*)d_in[20];
    const float* p_W1 = (const float*)d_in[21];  const float* p_b1 = (const float*)d_in[22];
    const float* p_g1 = (const float*)d_in[23];  const float* p_be1 = (const float*)d_in[24];
    const float* p_W2 = (const float*)d_in[25];  const float* p_b2 = (const float*)d_in[26];
    const float* p_gn = (const float*)d_in[27];  const float* p_bn = (const float*)d_in[28];

    float* out = (float*)d_out;
    float* nodesOut = out + OUT_NODES_OFF;       // 700 x 512

    // bump allocator over workspace
    size_t off = 0;
    auto alloc = [&](size_t bytes) -> void* {
        off = (off + 255) & ~(size_t)255;
        void* p = (char*)d_ws + off;
        off += bytes;
        return p;
    };
    float* agg   = (float*)alloc((size_t)N * HID * 4);
    float* cnt   = (float*)alloc(1024 * 4);
    float* mean  = (float*)alloc((size_t)N * HID * 4);
    float* h     = (float*)alloc((size_t)N * HID * 4);
    float* r1    = (float*)alloc((size_t)B * 800 * 4);
    float* t1    = (float*)alloc((size_t)B * 800 * 4);
    float* r2    = (float*)alloc((size_t)B * 1000 * 4);
    float* t2    = (float*)alloc((size_t)B * 1000 * 4);
    float* r3    = (float*)alloc((size_t)B * EMB * 4);
    float* imgf  = (float*)alloc((size_t)B * EMB * 4);
    float* A1    = (float*)alloc((size_t)NATTRS * 1000 * 4);
    float* O1    = (float*)alloc((size_t)NOBJS * 1000 * 4);
    unsigned short* pairN   = (unsigned short*)alloc((size_t)100000 * 512 * 2);
    // B operands (bf16, transposed, K-padded)
    unsigned short* s1_Wlt  = (unsigned short*)alloc((size_t)HID * 512 * 2);
    unsigned short* s1_Wrt  = (unsigned short*)alloc((size_t)HID * 512 * 2);
    unsigned short* s2_Wlt  = (unsigned short*)alloc((size_t)512 * 2048 * 2);
    unsigned short* s2_Wrt  = (unsigned short*)alloc((size_t)512 * 2048 * 2);
    unsigned short* i_W1t   = (unsigned short*)alloc((size_t)800 * 512 * 2);
    unsigned short* i_W2t   = (unsigned short*)alloc((size_t)1000 * 800 * 2);
    unsigned short* i_W3t   = (unsigned short*)alloc((size_t)512 * 1024 * 2);
    unsigned short* pW1tT   = (unsigned short*)alloc((size_t)1000 * 512 * 2);
    unsigned short* pW1tB   = (unsigned short*)alloc((size_t)1000 * 512 * 2);
    unsigned short* p_W2t   = (unsigned short*)alloc((size_t)512 * 1024 * 2);
    // A operands (bf16, row-major, M- and K-padded for unguarded async tiles)
    unsigned short* meanBf  = (unsigned short*)alloc((size_t)704 * 2048 * 2);
    unsigned short* nfBf    = (unsigned short*)alloc((size_t)704 * 512 * 2);
    unsigned short* hBf     = (unsigned short*)alloc((size_t)704 * 2048 * 2);
    unsigned short* imgBf   = (unsigned short*)alloc((size_t)256 * 512 * 2);
    unsigned short* t1Bf    = (unsigned short*)alloc((size_t)256 * 800 * 2);
    unsigned short* t2Bf    = (unsigned short*)alloc((size_t)256 * 1024 * 2);
    unsigned short* imgfBf  = (unsigned short*)alloc((size_t)256 * 512 * 2);
    unsigned short* nodesBf = (unsigned short*)alloc((size_t)768 * 512 * 2);

    dim3 blk(256);
    auto g1d = [&](long n) { return dim3((unsigned)cdiv(n, 256)); };

    // weight transpose+bf16 conversions
    tconv<<<g1d((long)2048 * 512), blk, 0, stream>>>(s1_Wl, s1_Wlt, 512, 2048, 2048, 512);
    tconv<<<g1d((long)2048 * 512), blk, 0, stream>>>(s1_Wr, s1_Wrt, 512, 2048, 2048, 512);
    tconv<<<g1d((long)512 * 2048), blk, 0, stream>>>(s2_Wl, s2_Wlt, 2048, 512, 512, 2048);
    tconv<<<g1d((long)512 * 2048), blk, 0, stream>>>(s2_Wr, s2_Wrt, 2048, 512, 512, 2048);
    tconv<<<g1d((long)800 * 512),  blk, 0, stream>>>(i_W1, i_W1t, 512, 800, 800, 512);
    tconv<<<g1d((long)1000 * 800), blk, 0, stream>>>(i_W2, i_W2t, 800, 1000, 1000, 800);
    tconv<<<g1d((long)512 * 1024), blk, 0, stream>>>(i_W3, i_W3t, 1000, 512, 512, 1024);
    tconv<<<g1d((long)1000 * 512), blk, 0, stream>>>(p_W1, pW1tT, 512, 1000, 1000, 512);
    tconv<<<g1d((long)1000 * 512), blk, 0, stream>>>(p_W1 + (size_t)512 * 1000, pW1tB, 512, 1000, 1000, 512);
    tconv<<<g1d((long)512 * 1024), blk, 0, stream>>>(p_W2, p_W2t, 1000, 512, 512, 1024);

    // ---- SAGE layer 1 ----
    zero_f32<<<g1d((long)N * 512), blk, 0, stream>>>(agg, N * 512);
    zero_f32<<<g1d(1024), blk, 0, stream>>>(cnt, 1024);
    seg_count<<<g1d(E), blk, 0, stream>>>(dst, cnt, E);
    seg_accum<<<g1d((long)E * 512), blk, 0, stream>>>(node_feats, src, dst, agg, E, 512);
    seg_mean<<<g1d((long)N * 512), blk, 0, stream>>>(agg, cnt, mean, N * 512, 512);
    convA<<<g1d((long)704 * 512), blk, 0, stream>>>(mean, meanBf, N, 704, 512, 512);
    convA<<<g1d((long)704 * 512), blk, 0, stream>>>(node_feats, nfBf, N, 704, 512, 512);
    {   // h = relu(mean@Wl + bl + x@Wr)
        dim3 grd(cdiv(N, 64), cdiv(HID, 128));
        gemm_bf16t<false, false, true><<<grd, blk, 0, stream>>>(meanBf, s1_Wlt, s1_bl, h, N, HID, 512, HID);
        gemm_bf16t<true, true, false><<<grd, blk, 0, stream>>>(nfBf, s1_Wrt, nullptr, h, N, HID, 512, HID);
    }
    // ---- SAGE layer 2 ----
    zero_f32<<<g1d((long)N * HID), blk, 0, stream>>>(agg, N * HID);
    seg_accum<<<g1d((long)E * HID), blk, 0, stream>>>(h, src, dst, agg, E, HID);
    seg_mean<<<g1d((long)N * HID), blk, 0, stream>>>(agg, cnt, mean, N * HID, HID);
    convA<<<g1d((long)704 * 2048), blk, 0, stream>>>(mean, meanBf, N, 704, 2048, 2048);
    convA<<<g1d((long)704 * 2048), blk, 0, stream>>>(h, hBf, N, 704, 2048, 2048);
    {   // nodes = mean@Wl + bl + h@Wr  (written straight into d_out tail)
        dim3 grd(cdiv(N, 64), cdiv(NODE_D, 128));
        gemm_bf16t<false, false, true><<<grd, blk, 0, stream>>>(meanBf, s2_Wlt, s2_bl, nodesOut, N, NODE_D, 2048, NODE_D);
        gemm_bf16t<true, false, false><<<grd, blk, 0, stream>>>(hBf, s2_Wrt, nullptr, nodesOut, N, NODE_D, 2048, NODE_D);
    }
    // ---- image MLP ----
    {
        convA<<<g1d((long)256 * 512), blk, 0, stream>>>(img, imgBf, B, 256, 512, 512);
        dim3 grd(cdiv(B, 64), cdiv(800, 128));
        gemm_bf16t<false, false, true><<<grd, blk, 0, stream>>>(imgBf, i_W1t, i_b1, r1, B, 800, 512, 800);
        ln_rows<<<g1d((long)B * 32), blk, 0, stream>>>(r1, i_g1, i_be1, t1, B, 800, 1);
        convA<<<g1d((long)256 * 800), blk, 0, stream>>>(t1, t1Bf, B, 256, 800, 800);
        dim3 grd2(cdiv(B, 64), cdiv(1000, 128));
        gemm_bf16t<false, false, true><<<grd2, blk, 0, stream>>>(t1Bf, i_W2t, i_b2, r2, B, 1000, 800, 1000);
        ln_rows<<<g1d((long)B * 32), blk, 0, stream>>>(r2, i_g2, i_be2, t2, B, 1000, 1);
        convA<<<g1d((long)256 * 1024), blk, 0, stream>>>(t2, t2Bf, B, 256, 1000, 1024);
        dim3 grd3(cdiv(B, 64), cdiv(EMB, 128));
        gemm_bf16t<false, false, true><<<grd3, blk, 0, stream>>>(t2Bf, i_W3t, i_b3, r3, B, EMB, 1024, EMB);
        ln_rows<<<g1d((long)B * 32), blk, 0, stream>>>(r3, i_gn, i_bn, imgf, B, EMB, 0);
        convA<<<g1d((long)256 * 512), blk, 0, stream>>>(imgf, imgfBf, B, 256, 512, 512);
    }
    // ---- pair network (factored) ----
    convA<<<g1d((long)768 * 512), blk, 0, stream>>>(nodesOut, nodesBf, N, 768, 512, 512);
    {   // A1 = attr@W1_top + b1 ; O1 = obj@W1_bot
        dim3 ga(cdiv(NATTRS, 64), cdiv(1000, 128));
        gemm_bf16t<false, false, true><<<ga, blk, 0, stream>>>(nodesBf, pW1tT, p_b1, A1, NATTRS, 1000, 512, 1000);
        dim3 go(cdiv(NOBJS, 64), cdiv(1000, 128));
        gemm_bf16t<false, false, false><<<go, blk, 0, stream>>>(nodesBf + (size_t)NATTRS * 512, pW1tB, nullptr, O1, NOBJS, 1000, 512, 1000);
    }
    pair_fused<<<dim3(100000 / 16), blk, 0, stream>>>(A1, O1, p_g1, p_be1, p_W2t, p_b2, p_gn, p_bn, pairN);

    // ---- pair_pred = img_feats @ pair_nodes^T ----
    {
        dim3 grd(cdiv(B, 64), cdiv(100000, 128));
        gemm_bf16t<false, false, false><<<grd, blk, 0, stream>>>(imgfBf, pairN, nullptr, out, B, 100000, 512, 100000);
    }
}
